// MultiHeadDaubechiesBlock_35948876268041
// MI455X (gfx1250) — matrix operand order repaired and checked by reading, not compiled, not run on hardware
//
#include <hip/hip_runtime.h>

typedef __bf16 bf16;
typedef __attribute__((ext_vector_type(16))) __bf16 v16bf;
typedef __attribute__((ext_vector_type(8)))  __bf16 v8bf;
typedef __attribute__((ext_vector_type(8)))  float  v8f;

static constexpr int Bb = 8, Tt = 4096, Dd = 512, Hh = 4, DHh = 128, FF = 2048;
static constexpr int Mrows = Bb * Tt;  // 32768

__device__ __forceinline__ bf16 f2bf(float f) {
  unsigned u = __builtin_bit_cast(unsigned, f);
  unsigned short h = (unsigned short)((u + 0x7fffu + ((u >> 16) & 1u)) >> 16);
  return __builtin_bit_cast(bf16, h);
}

// ---------------- weight conversion ----------------
__global__ __launch_bounds__(256) void cvt_f32_bf16(const float* __restrict__ in,
                                                    bf16* __restrict__ out, int n) {
  int i = blockIdx.x * 256 + threadIdx.x;
  if (i < n) out[i] = f2bf(in[i]);
}

// ---------------- LayerNorm 1: [B*T, D] -> channel-major fp32 [D][B*T] ----------------
// Tiled through LDS: coalesced global reads, per-row wave32 shuffle reductions,
// channel-major writes in contiguous 32-float (128 B) bursts.
__global__ __launch_bounds__(256) void ln1_to_cm(const float* __restrict__ x,
                                                 const float* __restrict__ g,
                                                 const float* __restrict__ be,
                                                 float* __restrict__ out_cm) {
  constexpr int RS = 513;                 // padded row stride (bank-conflict free columns)
  __shared__ float sx[32 * RS];
  __shared__ float smean[32], sinv[32];
  int r0 = blockIdx.x * 32;
  int tid = threadIdx.x;

  // load 32 rows x 512 cols, coalesced
  #pragma unroll 4
  for (int j = 0; j < 64; j++) {
    int flat = j * 256 + tid;
    int row = flat >> 9, col = flat & 511;
    sx[row * RS + col] = x[(size_t)(r0 + row) * Dd + col];
  }
  __syncthreads();

  // per-row reduction: wave w handles rows w, w+8, w+16, w+24
  int wave = tid >> 5, lane = tid & 31;
  for (int rr = wave; rr < 32; rr += 8) {
    float s = 0.f, q = 0.f;
    #pragma unroll
    for (int j = 0; j < 16; j++) {
      float v = sx[rr * RS + lane + 32 * j];
      s += v; q += v * v;
    }
    #pragma unroll
    for (int off = 16; off > 0; off >>= 1) {
      s += __shfl_xor(s, off);
      q += __shfl_xor(q, off);
    }
    if (lane == 0) {
      float mean = s * (1.0f / Dd);
      float var  = q * (1.0f / Dd) - mean * mean;
      smean[rr] = mean;
      sinv[rr]  = rsqrtf(var + 1e-5f);
    }
  }
  __syncthreads();

  // write channel-major: for each d, 32 consecutive rows -> contiguous 128 B
  #pragma unroll 4
  for (int j = 0; j < 64; j++) {
    int flat = j * 256 + tid;
    int d = flat >> 5, i = flat & 31;
    float v = (sx[i * RS + d] - smean[i]) * sinv[i] * g[d] + be[d];
    out_cm[(size_t)d * Mrows + r0 + i] = v;
  }
}

// ---------------- LayerNorm 2: [B*T, D] fp32 -> row-major bf16 ----------------
__global__ __launch_bounds__(256) void ln2_rm(const float* __restrict__ x,
                                              const float* __restrict__ g,
                                              const float* __restrict__ be,
                                              bf16* __restrict__ out) {
  int row = blockIdx.x;
  const float* xr = x + (size_t)row * Dd;
  int tid = threadIdx.x;
  float v0 = xr[tid], v1 = xr[tid + 256];
  __shared__ float s_sum[256], s_sq[256];
  s_sum[tid] = v0 + v1;
  s_sq[tid]  = v0 * v0 + v1 * v1;
  __syncthreads();
  for (int s = 128; s > 0; s >>= 1) {
    if (tid < s) { s_sum[tid] += s_sum[tid + s]; s_sq[tid] += s_sq[tid + s]; }
    __syncthreads();
  }
  float mean = s_sum[0] * (1.0f / Dd);
  float var  = s_sq[0] * (1.0f / Dd) - mean * mean;
  float inv  = rsqrtf(var + 1e-5f);
  float y0 = (v0 - mean) * inv * g[tid]       + be[tid];
  float y1 = (v1 - mean) * inv * g[tid + 256] + be[tid + 256];
  out[(size_t)row * Dd + tid]       = f2bf(y0);
  out[(size_t)row * Dd + tid + 256] = f2bf(y1);
}

// ---------------- DWT (3 levels) + linear interp + combine, per channel ----------------
__device__ __forceinline__ float interp1(const float* __restrict__ s, int L, int t) {
  float scale = (float)L * (1.0f / 4096.0f);
  float src = fmaxf(((float)t + 0.5f) * scale - 0.5f, 0.0f);
  int i0 = (int)floorf(src);
  if (i0 > L - 1) i0 = L - 1;
  int i1 = i0 + 1; if (i1 > L - 1) i1 = L - 1;
  float w = src - (float)i0;
  return s[i0] * (1.0f - w) + s[i1] * w;
}

__global__ __launch_bounds__(256) void dwt_combine(const float* __restrict__ xn_cm,
                                                   const float* __restrict__ h0,
                                                   const float* __restrict__ h1,
                                                   bf16* __restrict__ comb_cm) {
  int c = blockIdx.x;            // b*512 + d
  int b = c >> 9, d = c & 511;
  int h = d >> 7, dh = d & 127;
  int tid = threadIdx.x;
  const float* src = xn_cm + (size_t)d * Mrows + (size_t)b * Tt;
  bf16* dst = comb_cm + (size_t)d * Mrows + (size_t)b * Tt;

  __shared__ float a0[4096];
  __shared__ float a1[2048];
  __shared__ float a2[1024];
  __shared__ float a3[512];
  __shared__ float e0[2048];
  __shared__ float e1[1024];
  __shared__ float e2[512];

  for (int i = tid; i < 4096; i += 256) a0[i] = src[i];
  __syncthreads();

  float f0[4], f1[4];
  // level 0: L=4096 -> nw=2047
  #pragma unroll
  for (int k = 0; k < 4; k++) { int fi = ((0 * Hh + h) * 4 + k) * DHh + dh; f0[k] = h0[fi]; f1[k] = h1[fi]; }
  for (int w = tid; w < 2047; w += 256) {
    float s0 = 0.f, s1 = 0.f;
    #pragma unroll
    for (int k = 0; k < 4; k++) { float v = a0[2 * w + k]; s0 += f0[k] * v; s1 += f1[k] * v; }
    a1[w] = s0; e0[w] = s1;
  }
  __syncthreads();
  // level 1: L=2047 (zero-pad to 2048) -> nw=1023
  #pragma unroll
  for (int k = 0; k < 4; k++) { int fi = ((1 * Hh + h) * 4 + k) * DHh + dh; f0[k] = h0[fi]; f1[k] = h1[fi]; }
  for (int w = tid; w < 1023; w += 256) {
    float s0 = 0.f, s1 = 0.f;
    #pragma unroll
    for (int k = 0; k < 4; k++) { int idx = 2 * w + k; float v = (idx < 2047) ? a1[idx] : 0.0f; s0 += f0[k] * v; s1 += f1[k] * v; }
    a2[w] = s0; e1[w] = s1;
  }
  __syncthreads();
  // level 2: L=1023 (zero-pad to 1024) -> nw=511
  #pragma unroll
  for (int k = 0; k < 4; k++) { int fi = ((2 * Hh + h) * 4 + k) * DHh + dh; f0[k] = h0[fi]; f1[k] = h1[fi]; }
  for (int w = tid; w < 511; w += 256) {
    float s0 = 0.f, s1 = 0.f;
    #pragma unroll
    for (int k = 0; k < 4; k++) { int idx = 2 * w + k; float v = (idx < 1023) ? a2[idx] : 0.0f; s0 += f0[k] * v; s1 += f1[k] * v; }
    a3[w] = s0; e2[w] = s1;
  }
  __syncthreads();

  for (int t = tid; t < 4096; t += 256) {
    float v = interp1(e0, 2047, t) + interp1(e1, 1023, t)
            + interp1(e2, 511, t)  + interp1(a3, 511, t);
    dst[t] = f2bf(v);
  }
}

// ---------------- bf16 WMMA GEMM: C = A(MxK) * B(KxN) [+bias][+resid][gelu] ----------------
// block tile 128x128; 8 waves (2 M x 4 N), wave tile 64x32 -> 8 WMMAs per wave per K-step.
// A-fragment per ISA 16-bit 16x32 layout: lanes 0-15 hold K 0-7 & 16-23, lanes 16-31 hold K 8-15 & 24-31.
__device__ __forceinline__ v16bf fragA_ld(const bf16* base) {
  const v8bf* p = (const v8bf*)base;     // base = &As[row*40 + lhalf*8]
  v8bf lo = p[0];                        // K k0 .. k0+7
  v8bf hi = p[2];                        // K k0+16 .. k0+23
  v16bf r;
  #pragma unroll
  for (int i = 0; i < 8; i++) { r[i] = lo[i]; r[8 + i] = hi[i]; }
  return r;
}
// B-fragment: lane's column holds K 0-7 & 16-23 (lanes 0-15) or 8-15 & 24-31 (lanes 16-31), matching A.
__device__ __forceinline__ v16bf fragB_ld(const bf16* base) {
  const v8bf* p = (const v8bf*)base;     // base = &Bt[col*40 + lhalf*8]
  v8bf lo = p[0];
  v8bf hi = p[2];
  v16bf r;
  #pragma unroll
  for (int i = 0; i < 8; i++) { r[i] = lo[i]; r[8 + i] = hi[i]; }
  return r;
}

template<bool ATRANS, bool DO_GELU, bool DO_RESID, bool OUT_BF16>
__global__ __launch_bounds__(256) void gemm_bf16_wmma(
    const bf16* __restrict__ A, const bf16* __restrict__ Bw,
    const float* __restrict__ bias, const float* __restrict__ resid,
    float* __restrict__ Cf, bf16* __restrict__ Cb,
    int M, int N, int K) {
  __shared__ __align__(16) bf16 As[128 * 40];   // 128 rows (M) x 32 K, stride 40
  __shared__ __align__(16) bf16 Bt[128 * 40];   // 128 cols (N) x 32 K, stride 40 (K-contig per col)
  int tid = threadIdx.x;
  int lane = tid & 31, wave = tid >> 5;
  int wm = wave >> 2, wn = wave & 3;            // 2x4 wave grid -> 128x128 block tile
  int l16 = lane & 15, lhalf = lane >> 4;
  int m0 = blockIdx.y * 128, n0 = blockIdx.x * 128;

  unsigned asbase = (unsigned)(size_t)(&As[0]); // LDS byte offset (low 32 bits of flat addr)

  v8f acc[4][2];
  #pragma unroll
  for (int mi = 0; mi < 4; mi++)
    #pragma unroll
    for (int ni = 0; ni < 2; ni++) acc[mi][ni] = v8f{};

  for (int kb = 0; kb < K; kb += 32) {
    if constexpr (ATRANS) {                     // A stored [K][M] (channel-major)
      #pragma unroll
      for (int j = 0; j < 2; j++) {
        int s = tid + j * 256;
        int k = s >> 4, mp = (s & 15) * 8;
        v8bf v = *(const v8bf*)(A + (size_t)(kb + k) * M + m0 + mp);
        #pragma unroll
        for (int e = 0; e < 8; e++) As[(mp + e) * 40 + k] = v[e];
      }
    } else {                                    // A row-major [M][K]: async global->LDS copy
      #pragma unroll
      for (int j = 0; j < 2; j++) {
        int s = tid + j * 256;
        int r = s >> 2, kp = (s & 3) * 8;
        unsigned lds = asbase + (unsigned)((r * 40 + kp) * 2);
        unsigned long long ga =
            (unsigned long long)(size_t)(A + (size_t)(m0 + r) * K + kb + kp);
        asm volatile("global_load_async_to_lds_b128 %0, %1, off"
                     :: "v"(lds), "v"(ga) : "memory");
      }
    }
    {                                           // B row-major [K][N] -> LDS transposed
      #pragma unroll
      for (int j = 0; j < 2; j++) {
        int s = tid + j * 256;
        int k = s >> 4, np = (s & 15) * 8;
        v8bf v = *(const v8bf*)(Bw + (size_t)(kb + k) * N + n0 + np);
        #pragma unroll
        for (int e = 0; e < 8; e++) Bt[(np + e) * 40 + k] = v[e];
      }
    }
    if constexpr (!ATRANS) {
      asm volatile("s_wait_asynccnt 0x0" ::: "memory");
    }
    __syncthreads();

    v16bf af[4], bf_[2];
    #pragma unroll
    for (int mi = 0; mi < 4; mi++)
      af[mi] = fragA_ld(&As[(wm * 64 + mi * 16 + l16) * 40 + lhalf * 8]);
    #pragma unroll
    for (int ni = 0; ni < 2; ni++)
      bf_[ni] = fragB_ld(&Bt[(wn * 32 + ni * 16 + l16) * 40 + lhalf * 8]);

    #pragma unroll
    for (int mi = 0; mi < 4; mi++)
      #pragma unroll
      for (int ni = 0; ni < 2; ni++)
        acc[mi][ni] = __builtin_amdgcn_wmma_f32_16x16x32_bf16(
            false, af[mi], false, bf_[ni], (short)0, acc[mi][ni], false, false);
    __syncthreads();
  }

  // epilogue: C VGPR i holds M = i (lanes 0-15) or 8+i (lanes 16-31), N = lane%16
  #pragma unroll
  for (int mi = 0; mi < 4; mi++) {
    #pragma unroll
    for (int ni = 0; ni < 2; ni++) {
      int col = n0 + wn * 32 + ni * 16 + l16;
      float bv = bias[col];
      #pragma unroll
      for (int i = 0; i < 8; i++) {
        int row = m0 + wm * 64 + mi * 16 + lhalf * 8 + i;
        float v = acc[mi][ni][i] + bv;
        if constexpr (DO_RESID) v += resid[(size_t)row * N + col];
        if constexpr (DO_GELU)  v = 0.5f * v * (1.0f + erff(v * 0.70710678118654752f));
        if constexpr (OUT_BF16) Cb[(size_t)row * N + col] = f2bf(v);
        else                    Cf[(size_t)row * N + col] = v;
      }
    }
  }
}

// ---------------- launch ----------------
extern "C" void kernel_launch(void* const* d_in, const int* in_sizes, int n_in,
                              void* d_out, int out_size, void* d_ws, size_t ws_size,
                              hipStream_t stream) {
  (void)in_sizes; (void)n_in; (void)out_size; (void)ws_size;
  const float* x      = (const float*)d_in[0];
  const float* h0     = (const float*)d_in[1];
  const float* h1     = (const float*)d_in[2];
  const float* ln1_g  = (const float*)d_in[3];
  const float* ln1_b  = (const float*)d_in[4];
  const float* ln2_g  = (const float*)d_in[5];
  const float* ln2_b  = (const float*)d_in[6];
  const float* proj_w = (const float*)d_in[7];
  const float* proj_b = (const float*)d_in[8];
  const float* w1     = (const float*)d_in[9];
  const float* b1     = (const float*)d_in[10];
  const float* w2     = (const float*)d_in[11];
  const float* b2     = (const float*)d_in[12];
  float* out = (float*)d_out;
  char* ws = (char*)d_ws;

  float* xn_cm = (float*)(ws);                         // 64 MB  [D][B*T] fp32
  bf16*  comb  = (bf16*)(ws + 67108864ull);            // 32 MB  [D][B*T] bf16 (A^T for proj GEMM)
  float* x2    = (float*)(ws + 100663296ull);          // 64 MB  [B*T][D] fp32 (post-proj residual)
  bf16*  xn2   = (bf16*)(ws + 167772160ull);           // 32 MB  [B*T][D] bf16
  bf16*  hdn   = (bf16*)(ws + 201326592ull);           // 128 MB [B*T][FFN] bf16
  bf16*  pwb   = (bf16*)(ws + 335544320ull);           // 512 KB
  bf16*  w1b   = (bf16*)(ws + 336068608ull);           // 2 MB
  bf16*  w2b   = (bf16*)(ws + 338165760ull);           // 2 MB

  cvt_f32_bf16<<<(Dd * Dd) / 256, 256, 0, stream>>>(proj_w, pwb, Dd * Dd);
  cvt_f32_bf16<<<(Dd * FF) / 256, 256, 0, stream>>>(w1, w1b, Dd * FF);
  cvt_f32_bf16<<<(FF * Dd) / 256, 256, 0, stream>>>(w2, w2b, FF * Dd);

  ln1_to_cm<<<Mrows / 32, 256, 0, stream>>>(x, ln1_g, ln1_b, xn_cm);
  dwt_combine<<<Bb * Dd, 256, 0, stream>>>(xn_cm, h0, h1, comb);

  // x2 = x + (combined @ proj_w + proj_b)
  gemm_bf16_wmma<true, false, true, false><<<dim3(Dd / 128, Mrows / 128), 256, 0, stream>>>(
      comb, pwb, proj_b, x, x2, nullptr, Mrows, Dd, Dd);

  ln2_rm<<<Mrows, 256, 0, stream>>>(x2, ln2_g, ln2_b, xn2);

  // hdn = gelu(xn2 @ w1 + b1)
  gemm_bf16_wmma<false, true, false, true><<<dim3(FF / 128, Mrows / 128), 256, 0, stream>>>(
      xn2, w1b, b1, nullptr, nullptr, hdn, Mrows, FF, Dd);

  // out = x2 + (hdn @ w2 + b2)
  gemm_bf16_wmma<false, false, true, false><<<dim3(Dd / 128, Mrows / 128), 256, 0, stream>>>(
      hdn, w2b, b2, x2, out, nullptr, Mrows, Dd, FF);
}
